// GNN_43035572305985
// MI455X (gfx1250) — compile-verified, matrix operand order checked by reading, not run on hardware
//
#include <hip/hip_runtime.h>

// ---------------- problem dims (fixed by the reference) ----------------
static constexpr int N_   = 50000;   // nodes
static constexpr int E_   = 800000;  // edges
static constexpr int G_   = 512;     // graphs
static constexpr int H_   = 128;     // hidden dim
static constexpr int OUT_ = 64;      // output dim
static constexpr int WSZ  = H_ * H_; // 16384 weights per matrix

typedef __attribute__((ext_vector_type(16))) __bf16    v16bf;
typedef __attribute__((ext_vector_type(8)))  float     v8f;
typedef __attribute__((ext_vector_type(4)))  unsigned  u32x4;
typedef __attribute__((ext_vector_type(8)))  int       i32x8;
typedef __attribute__((ext_vector_type(4)))  int       i32x4;

static __device__ __forceinline__ unsigned short f2bf_bits(float f) {
  unsigned u = __float_as_uint(f);
  u += 0x7FFFu + ((u >> 16) & 1u);              // round-to-nearest-even
  return (unsigned short)(u >> 16);
}

// order-preserving float<->int encoding for atomicMax on floats
static __device__ __forceinline__ int   enc_f(float f) { int i = __float_as_int(f); return i >= 0 ? i : (i ^ 0x7FFFFFFF); }
static __device__ __forceinline__ float dec_f(int i)   { return __int_as_float(i >= 0 ? i : (i ^ 0x7FFFFFFF)); }

static __device__ __forceinline__ unsigned lds_off(const void* p) {
  return (unsigned)(unsigned long long)(size_t)p;   // low 32 bits = LDS byte offset
}

// ---- Tensor Data Mover: 2-D tile (rows x cols of `data_size` elems) -> LDS ----
// D# per CDNA5 ISA: group0 = {flags, lds_addr, global_addr, type=2},
// group1 = {mask/data_size, tensor_dim0/1, tile_dim0/1/2, strides}; groups 2/3 zero (2-D).
static __device__ __forceinline__ void tdm_load_2d(unsigned lds_addr, const void* gptr,
                                                   unsigned tdim0, unsigned tdim1,
                                                   unsigned tile0, unsigned tile1,
                                                   unsigned stride0, unsigned dsz_log) {
  unsigned long long ga = (unsigned long long)(size_t)gptr;
  u32x4 g0;
  g0[0] = 1u;                                               // count=1 (user descriptor)
  g0[1] = lds_addr;                                         // lds_addr (bytes)
  g0[2] = (unsigned)(ga & 0xFFFFFFFFu);                     // global_addr[31:0]
  g0[3] = (unsigned)((ga >> 32) & 0x01FFFFFFu) | 0x80000000u; // addr[56:32] | type=2
  i32x8 g1;
  g1[0] = (int)(dsz_log << 16);                             // wg_mask=0, data_size
  g1[1] = (int)((tdim0 & 0xFFFFu) << 16);                   // tensor_dim0[15:0]
  g1[2] = (int)(((tdim0 >> 16) & 0xFFFFu) | ((tdim1 & 0xFFFFu) << 16));
  g1[3] = (int)(((tdim1 >> 16) & 0xFFFFu) | ((tile0 & 0xFFFFu) << 16));
  g1[4] = (int)(tile1 & 0xFFFFu);                           // tile_dim1; tile_dim2=0
  g1[5] = (int)stride0;                                     // tensor_dim0_stride[31:0]
  g1[6] = 0;
  g1[7] = 0;
  i32x4 z4 = {0, 0, 0, 0};
#if __clang_major__ >= 23
  i32x8 z8 = {0, 0, 0, 0, 0, 0, 0, 0};
  __builtin_amdgcn_tensor_load_to_lds(g0, g1, z4, z4, z8, 0);
#else
  __builtin_amdgcn_tensor_load_to_lds(g0, g1, z4, z4, 0);
#endif
}

// ---- fp32 -> bf16 streaming conversion ----
__global__ void k_cvt_bf16(const float* __restrict__ X, unsigned short* __restrict__ Xb) {
  size_t i = (size_t)blockIdx.x * blockDim.x + threadIdx.x;
  Xb[i] = f2bf_bits(X[i]);
}

// ---- pack one 128x128 fp32 weight matrix into bf16 WMMA-fragment order ----
// layout: Wp[(((ntile*4)+kb)*32 + lane)*16 + i], so each lane reads 32 contiguous bytes.
__global__ void k_packW(const float* __restrict__ W, unsigned short* __restrict__ Wp) {
  int t = blockIdx.x * blockDim.x + threadIdx.x;   // 0..16383
  int i    = t & 15;
  int lane = (t >> 4) & 31;
  int kb   = (t >> 9) & 3;
  int nt   = t >> 11;
  int hi   = lane >> 4;
  int k = kb * 32 + hi * 8 + (i & 7) + ((i >> 3) << 4);
  Wp[t] = f2bf_bits(W[k * H_ + nt * 16 + (lane & 15)]);
}

// ============== WMMA GEMM: Y[rows,128] (+)= Xb[rows,128](bf16) @ Wp(bf16 packed) ==============
// block = 256 threads = 8 waves; TDM stages the 16x128 A-tile (4KB) and packed W (32KB) in LDS,
// then the inner loop is ds_load_b128 + v_wmma_f32_16x16x32_bf16 only.
__global__ __launch_bounds__(256) void k_gemm128(const unsigned short* __restrict__ Xb,
                                                 const unsigned short* __restrict__ Wp,
                                                 float* __restrict__ Y, int accum) {
  __shared__ __align__(16) unsigned short aTile[16 * H_];      // 4 KB
  __shared__ __align__(16) unsigned short wTile[WSZ];          // 32 KB

  const int tid  = threadIdx.x;
  const int lane = tid & 31;
  const int wave = tid >> 5;
  const int m0   = blockIdx.x << 4;
  const int n0   = wave << 4;
  const int mr   = lane & 15;
  const int hi   = lane >> 4;

  if (tid == 0) {
    tdm_load_2d(lds_off(aTile), Xb + (size_t)m0 * H_, H_, 16, H_, 16, H_, 1);
    tdm_load_2d(lds_off(wTile), Wp, WSZ, 1, WSZ, 1, WSZ, 1);
    __builtin_amdgcn_s_wait_tensorcnt(0);
  }
  __syncthreads();

  v8f c = {0.f, 0.f, 0.f, 0.f, 0.f, 0.f, 0.f, 0.f};
  if (accum) {
#pragma unroll
    for (int r = 0; r < 8; ++r)
      c[r] = Y[(size_t)(m0 + r + (hi << 3)) * H_ + n0 + mr];
  }

#pragma unroll
  for (int kb = 0; kb < 4; ++kb) {
    union { uint4 q[2]; v16bf v; } A, B;
    const int abase = mr * H_ + kb * 32 + hi * 8;          // elems; 16B aligned
    A.q[0] = *(const uint4*)&aTile[abase];                 // K = kb*32+hi*8 .. +7
    A.q[1] = *(const uint4*)&aTile[abase + 16];            // K = +16 .. +23
    const int wbase = ((wave * 4 + kb) * 32 + lane) * 16;  // 32B contiguous per lane
    B.q[0] = *(const uint4*)&wTile[wbase];
    B.q[1] = *(const uint4*)&wTile[wbase + 8];
    c = __builtin_amdgcn_wmma_f32_16x16x32_bf16(false, A.v, false, B.v, (short)0, c,
                                                false, false);
  }

#pragma unroll
  for (int r = 0; r < 8; ++r)
    Y[(size_t)(m0 + r + (hi << 3)) * H_ + n0 + mr] = c[r];
}

// ============== graph / edge kernels ==============
__global__ void k_deg(const int* __restrict__ dst, float* __restrict__ deg) {
  int e = blockIdx.x * blockDim.x + threadIdx.x;
  atomicAdd(&deg[dst[e]], 1.0f);
}

__global__ void k_dis(float* __restrict__ deg) {
  int i = blockIdx.x * blockDim.x + threadIdx.x;
  if (i < N_) {
    float d = deg[i];
    deg[i] = d > 0.f ? rsqrtf(fmaxf(d, 1.f)) : 0.f;   // in place: deg -> D^-1/2
  }
}

__global__ void k_norm(const int* __restrict__ src, const int* __restrict__ dst,
                       const float* __restrict__ dis, float* __restrict__ norm) {
  int e = blockIdx.x * blockDim.x + threadIdx.x;
  norm[e] = dis[src[e]] * dis[dst[e]];
}

__global__ void k_scale_copy(float* __restrict__ y, const float* __restrict__ x, float a) {
  size_t i = (size_t)blockIdx.x * blockDim.x + threadIdx.x;
  y[i] = a * x[i];
}

// y[dst] += alpha * norm[e] * x[src]  (32 lanes per edge, float4 per lane)
__global__ void k_spmm(float* __restrict__ y, const float* __restrict__ x,
                       const float* __restrict__ w, const int* __restrict__ src,
                       const int* __restrict__ dst, float alpha) {
  int gid = blockIdx.x * blockDim.x + threadIdx.x;
  int e = gid >> 5, lane = gid & 31;
  float ww = alpha * w[e];
  int s = src[e], d = dst[e];
  float4 xv = reinterpret_cast<const float4*>(x)[(size_t)s * 32 + lane];
  float* yb = y + (size_t)d * H_ + lane * 4;
  atomicAdd(yb + 0, ww * xv.x);
  atomicAdd(yb + 1, ww * xv.y);
  atomicAdd(yb + 2, ww * xv.z);
  atomicAdd(yb + 3, ww * xv.w);
}

__global__ void k_bias_act(float* __restrict__ dst_, const float* __restrict__ src_,
                           const float* __restrict__ bias, int relu) {
  size_t i = (size_t)blockIdx.x * blockDim.x + threadIdx.x;
  float v = src_[i] + bias[i & (H_ - 1)];
  if (relu) v = fmaxf(v, 0.f);
  dst_[i] = v;
}

__global__ void k_add2_relu(float* __restrict__ dst_, const float* __restrict__ a,
                            const float* __restrict__ b) {
  size_t i = (size_t)blockIdx.x * blockDim.x + threadIdx.x;
  dst_[i] = fmaxf(a[i] + b[i], 0.f);
}

__global__ void k_initm(int* __restrict__ m) {
  int i = blockIdx.x * blockDim.x + threadIdx.x;
  if (i < N_) m[i] = (int)0x807FFFFF;  // enc(-inf)
}

// score[e] = <q[dst], k[src]> / sqrt(H); segment max via ordered-int atomicMax
__global__ void k_score(const float* __restrict__ q, const float* __restrict__ kk,
                        const int* __restrict__ src, const int* __restrict__ dst,
                        float* __restrict__ score, int* __restrict__ mbits) {
  int gid = blockIdx.x * blockDim.x + threadIdx.x;
  int e = gid >> 5, lane = gid & 31;
  int s = src[e], d = dst[e];
  float4 qa = reinterpret_cast<const float4*>(q)[(size_t)d * 32 + lane];
  float4 kb = reinterpret_cast<const float4*>(kk)[(size_t)s * 32 + lane];
  float p = qa.x * kb.x + qa.y * kb.y + qa.z * kb.z + qa.w * kb.w;
#pragma unroll
  for (int off = 16; off > 0; off >>= 1) p += __shfl_xor(p, off, 32);
  if (lane == 0) {
    float sc = p * 0.08838834764831845f;  // 1/sqrt(128)
    score[e] = sc;
    atomicMax(&mbits[d], enc_f(sc));
  }
}

__global__ void k_expden(const float* __restrict__ score, float* __restrict__ eexp,
                         float* __restrict__ denom, const int* __restrict__ mbits,
                         const int* __restrict__ dst) {
  int e = blockIdx.x * blockDim.x + threadIdx.x;
  int d = dst[e];
  float m = dec_f(mbits[d]);
  float ex = expf(score[e] - m);
  eexp[e] = ex;
  atomicAdd(&denom[d], ex);
}

__global__ void k_agg(float* __restrict__ agg, const float* __restrict__ v,
                      const float* __restrict__ eexp, const float* __restrict__ denom,
                      const int* __restrict__ src, const int* __restrict__ dst) {
  int gid = blockIdx.x * blockDim.x + threadIdx.x;
  int e = gid >> 5, lane = gid & 31;
  int s = src[e], d = dst[e];
  float al = eexp[e] / (denom[d] + 1e-16f);
  float4 vv = reinterpret_cast<const float4*>(v)[(size_t)s * 32 + lane];
  float* yb = agg + (size_t)d * H_ + lane * 4;
  atomicAdd(yb + 0, al * vv.x);
  atomicAdd(yb + 1, al * vv.y);
  atomicAdd(yb + 2, al * vv.z);
  atomicAdd(yb + 3, al * vv.w);
}

__global__ void k_pool(const float* __restrict__ h, const int* __restrict__ batch,
                       float* __restrict__ sums, float* __restrict__ cnt) {
  int gid = blockIdx.x * blockDim.x + threadIdx.x;
  int n = gid >> 5, lane = gid & 31;
  int g = batch[n];
  if (lane == 0) atomicAdd(&cnt[g], 1.0f);
  float4 hv = reinterpret_cast<const float4*>(h)[(size_t)n * 32 + lane];
  float* sb = sums + (size_t)g * H_ + lane * 4;
  atomicAdd(sb + 0, hv.x);
  atomicAdd(sb + 1, hv.y);
  atomicAdd(sb + 2, hv.z);
  atomicAdd(sb + 3, hv.w);
}

// z1[g][o] = b1[o] + concat(h[root[g]], pool[g]) . W1[:,o]
__global__ void k_z1(const float* __restrict__ h, const int* __restrict__ root,
                     const float* __restrict__ sums, const float* __restrict__ cnt,
                     const float* __restrict__ W1, const float* __restrict__ b1,
                     float* __restrict__ z1) {
  int gid = blockIdx.x * blockDim.x + threadIdx.x;  // G*OUT
  int g = gid >> 6, o = gid & 63;
  int r = root[g];
  float inv = 1.0f / fmaxf(cnt[g], 1.0f);
  float acc = b1[o];
  const float* hr = h + (size_t)r * H_;
  const float* sg = sums + (size_t)g * H_;
  for (int j = 0; j < H_; ++j) {
    acc += hr[j] * W1[j * OUT_ + o];
    acc += sg[j] * inv * W1[(H_ + j) * OUT_ + o];
  }
  z1[gid] = acc;
}

// per-column softmax over G=512 entries, then mean over axis 0 -> msm[j]
__global__ void k_softmax_mean(const float* __restrict__ z, float* __restrict__ msm) {
  __shared__ float red[256];
  int j = blockIdx.x, t = threadIdx.x;
  float v0 = z[t * OUT_ + j];
  float v1 = z[(t + 256) * OUT_ + j];
  red[t] = fmaxf(v0, v1);
  __syncthreads();
  for (int s = 128; s > 0; s >>= 1) { if (t < s) red[t] = fmaxf(red[t], red[t + s]); __syncthreads(); }
  float m = red[0];
  __syncthreads();
  float e = expf(v0 - m) + expf(v1 - m);
  red[t] = e;
  __syncthreads();
  for (int s = 128; s > 0; s >>= 1) { if (t < s) red[t] += red[t + s]; __syncthreads(); }
  float den = red[0];
  __syncthreads();
  red[t] = e / den;
  __syncthreads();
  for (int s = 128; s > 0; s >>= 1) { if (t < s) red[t] += red[t + s]; __syncthreads(); }
  if (t == 0) msm[j] = red[0] / (float)G_;
}

// out[o] = b2[o] + msm . W2[:,o]   (mean and the final linear commute exactly)
__global__ void k_final(const float* __restrict__ msm, const float* __restrict__ W2,
                        const float* __restrict__ b2, float* __restrict__ out) {
  int o = threadIdx.x;
  float acc = b2[o];
  for (int j = 0; j < OUT_; ++j) acc += msm[j] * W2[j * OUT_ + o];
  out[o] = acc;
}

extern "C" void kernel_launch(void* const* d_in, const int* in_sizes, int n_in,
                              void* d_out, int out_size, void* d_ws, size_t ws_size,
                              hipStream_t stream) {
  (void)in_sizes; (void)n_in; (void)out_size; (void)ws_size;
  const float* x    = (const float*)d_in[0];
  const int*   esrc = (const int*)d_in[1];
  const int*   edst = (const int*)d_in[2];
  const int*   batch= (const int*)d_in[3];
  const int*   root = (const int*)d_in[4];
  const float* Wc   = (const float*)d_in[5];
  const float* bc   = (const float*)d_in[6];
  const float* Wt   = (const float*)d_in[7];
  const float* bt   = (const float*)d_in[8];
  const float* Wq   = (const float*)d_in[9];
  const float* bq   = (const float*)d_in[10];
  const float* Wk   = (const float*)d_in[11];
  const float* bk   = (const float*)d_in[12];
  const float* Wv   = (const float*)d_in[13];
  const float* bv   = (const float*)d_in[14];
  const float* Wsm  = (const float*)d_in[15];
  const float* bs   = (const float*)d_in[16];
  const float* W1   = (const float*)d_in[17];
  const float* b1   = (const float*)d_in[18];
  const float* W2   = (const float*)d_in[19];
  const float* b2   = (const float*)d_in[20];
  float* out = (float*)d_out;

  // ---- workspace carve-up (~180 MB) ----
  float* ws = (float*)d_ws;
  size_t off = 0;
  auto take = [&](size_t n) { float* p = ws + off; off += n; return p; };
  float* deg   = take(N_);               // becomes D^-1/2 in place
  float* norm  = take(E_);
  float* score = take(E_);
  float* eexp  = take(E_);
  int*   mbits = (int*)take(N_);
  float* denom = take(N_);
  float* cnt   = take(G_);
  float* sums  = take((size_t)G_ * H_);
  float* z1    = take((size_t)G_ * OUT_);
  float* msm   = take(OUT_);
  float* B0 = take((size_t)N_ * H_);     // GEMM accumulator
  float* B1 = take((size_t)N_ * H_);
  float* B2 = take((size_t)N_ * H_);
  float* B3 = take((size_t)N_ * H_);
  float* B4 = take((size_t)N_ * H_);     // h
  float* B5 = take((size_t)N_ * H_);     // agg
  unsigned short* Xbf = (unsigned short*)take((size_t)N_ * H_ / 2);  // bf16 activations
  unsigned short* Wp  = (unsigned short*)take((size_t)13 * WSZ / 2); // 13 packed weights

  const int TPB = 256;
  const int gE  = E_ / TPB;              // 3125
  const int gEw = (E_ * 32) / TPB;       // 100000 (wave per edge)
  const int gNH = (N_ * H_) / TPB;       // 25000
  const int gNg = (N_ + TPB - 1) / TPB;  // 196
  const int gNw = (N_ * 32) / TPB;       // 6250 (wave per node)
  const int gMM = N_ / 16;               // 3125 row-tiles
  const size_t NHb = (size_t)N_ * H_ * sizeof(float);

  // GEMM helper: cvt input to bf16 then WMMA
  auto gemm = [&](const float* X, int widx, float* Y, int acc) {
    k_cvt_bf16<<<gNH, TPB, 0, stream>>>(X, Xbf);
    k_gemm128<<<gMM, TPB, 0, stream>>>(Xbf, Wp + (size_t)widx * WSZ, Y, acc);
  };

  // ---- pack all 13 weight matrices once ----
  const float* Wlist[13] = {Wc, Wc + WSZ, Wc + 2 * WSZ, Wc + 3 * WSZ, Wc + 4 * WSZ,
                            Wt, Wt + WSZ, Wt + 2 * WSZ, Wt + 3 * WSZ,
                            Wq, Wk, Wv, Wsm};
  for (int i = 0; i < 13; ++i)
    k_packW<<<WSZ / TPB, TPB, 0, stream>>>(Wlist[i], Wp + (size_t)i * WSZ);

  // ---- degrees + symmetric norm ----
  hipMemsetAsync(deg, 0, N_ * sizeof(float), stream);
  k_deg<<<gE, TPB, 0, stream>>>(edst, deg);
  k_dis<<<gNg, TPB, 0, stream>>>(deg);
  k_norm<<<gE, TPB, 0, stream>>>(esrc, edst, deg, norm);

  // ---- ChebConv (K=5): w_cheb = -norm; ACC=B0, h -> B4 ----
  gemm(x, 0, B0, 0);
  hipMemsetAsync(B1, 0, NHb, stream);                                  // Tx1
  k_spmm<<<gEw, TPB, 0, stream>>>(B1, x, norm, esrc, edst, -1.0f);
  gemm(B1, 1, B0, 1);
  k_scale_copy<<<gNH, TPB, 0, stream>>>(B2, x, -1.0f);                 // Tx2 = -Tx0 ...
  k_spmm<<<gEw, TPB, 0, stream>>>(B2, B1, norm, esrc, edst, -2.0f);    // ... - 2*norm*Tx1
  gemm(B2, 2, B0, 1);
  k_scale_copy<<<gNH, TPB, 0, stream>>>(B3, B1, -1.0f);                // Tx3
  k_spmm<<<gEw, TPB, 0, stream>>>(B3, B2, norm, esrc, edst, -2.0f);
  gemm(B3, 3, B0, 1);
  k_scale_copy<<<gNH, TPB, 0, stream>>>(B1, B2, -1.0f);                // Tx4 (reuse B1)
  k_spmm<<<gEw, TPB, 0, stream>>>(B1, B3, norm, esrc, edst, -2.0f);
  gemm(B1, 4, B0, 1);
  k_bias_act<<<gNH, TPB, 0, stream>>>(B4, B0, bc, 1);                  // h = relu(out+bc)

  // ---- TAGConv (K=3): ACC=B0, h stays B4 ----
  gemm(B4, 5, B0, 0);
  hipMemsetAsync(B1, 0, NHb, stream);
  k_spmm<<<gEw, TPB, 0, stream>>>(B1, B4, norm, esrc, edst, 1.0f);
  gemm(B1, 6, B0, 1);
  hipMemsetAsync(B2, 0, NHb, stream);
  k_spmm<<<gEw, TPB, 0, stream>>>(B2, B1, norm, esrc, edst, 1.0f);
  gemm(B2, 7, B0, 1);
  hipMemsetAsync(B3, 0, NHb, stream);
  k_spmm<<<gEw, TPB, 0, stream>>>(B3, B2, norm, esrc, edst, 1.0f);
  gemm(B3, 8, B0, 1);
  k_bias_act<<<gNH, TPB, 0, stream>>>(B4, B0, bt, 1);                  // h = relu(out+bt)

  // ---- TransformerConv: q=B1, k=B2, v=B3, skip=B0, agg=B5 ----
  gemm(B4, 9, B1, 0);
  k_bias_act<<<gNH, TPB, 0, stream>>>(B1, B1, bq, 0);
  gemm(B4, 10, B2, 0);
  k_bias_act<<<gNH, TPB, 0, stream>>>(B2, B2, bk, 0);
  gemm(B4, 11, B3, 0);
  k_bias_act<<<gNH, TPB, 0, stream>>>(B3, B3, bv, 0);
  gemm(B4, 12, B0, 0);
  k_bias_act<<<gNH, TPB, 0, stream>>>(B0, B0, bs, 0);

  k_initm<<<gNg, TPB, 0, stream>>>(mbits);
  hipMemsetAsync(denom, 0, N_ * sizeof(float), stream);
  k_score<<<gEw, TPB, 0, stream>>>(B1, B2, esrc, edst, score, mbits);
  k_expden<<<gE, TPB, 0, stream>>>(score, eexp, denom, mbits, edst);
  hipMemsetAsync(B5, 0, NHb, stream);
  k_agg<<<gEw, TPB, 0, stream>>>(B5, B3, eexp, denom, esrc, edst);
  k_add2_relu<<<gNH, TPB, 0, stream>>>(B4, B5, B0);                    // h = relu(agg+skip)

  // ---- readout ----
  hipMemsetAsync(cnt, 0, G_ * sizeof(float), stream);
  hipMemsetAsync(sums, 0, (size_t)G_ * H_ * sizeof(float), stream);
  k_pool<<<gNw, TPB, 0, stream>>>(B4, batch, sums, cnt);
  k_z1<<<(G_ * OUT_) / TPB, TPB, 0, stream>>>(B4, root, sums, cnt, W1, b1, z1);
  k_softmax_mean<<<OUT_, 256, 0, stream>>>(z1, msm);
  k_final<<<1, OUT_, 0, stream>>>(msm, W2, b2, out);
}
